// TemporalAttentionLayer_28991029248521
// MI455X (gfx1250) — matrix operand order repaired and checked by reading, not compiled, not run on hardware
//
#include <hip/hip_runtime.h>
#include <hip/hip_bf16.h>

typedef __bf16 bf16;
typedef __attribute__((ext_vector_type(16))) __bf16 v16bf;
typedef __attribute__((ext_vector_type(8)))  __bf16 v8bf;
typedef __attribute__((ext_vector_type(4)))  __bf16 v4bf;
typedef __attribute__((ext_vector_type(8)))  float  v8f;

namespace {

constexpr int B_   = 100000;
constexpr int N_   = 20;
constexpr int F_   = 172;
constexpr int E_   = 344;   // embed dim
constexpr int EP_  = 352;   // E padded to mult of 16
constexpr int KD_  = 516;   // key/value input dim
constexpr int KDP_ = 544;   // KD padded to mult of 32
constexpr int HID_ = 172;
constexpr int HP_  = 176;   // HID padded to 16 (N dim of fc1)
constexpr int H2P_ = 192;   // HID padded to 32 (K dim of fc2)
constexpr float SCALE_ = 0.07624928516630233f; // 1/sqrt(172)

// ---- workspace byte offsets (all 64B aligned) ----
constexpr size_t WQ_OFF = 0;                                    // [352][352] bf16
constexpr size_t WK_OFF = WQ_OFF + (size_t)EP_ * EP_  * 2;      // [352][544] bf16
constexpr size_t WV_OFF = WK_OFF + (size_t)EP_ * KDP_ * 2;      // [352][544] bf16
constexpr size_t WO_OFF = WV_OFF + (size_t)EP_ * KDP_ * 2;      // [352][352] bf16
constexpr size_t W1_OFF = WO_OFF + (size_t)EP_ * EP_  * 2;      // [176][544] bf16
constexpr size_t W2_OFF = W1_OFF + (size_t)HP_ * KDP_ * 2;      // [176][192] bf16
constexpr size_t QW_OFF = W2_OFF + (size_t)HP_ * H2P_ * 2;      // [B][344] f32
// total workspace = QW_OFF + B_*E_*4  (~139 MB assumed available in d_ws)

// ---- LDS layout for k_attn (bytes). kvA region is reused after phase 1. ----
constexpr int KVA_OFF  = 0;                 // [160][544] bf16 = 174080
constexpr int CTXA_OFF = 0;                 // [16][352]  bf16 (aliases dead kvA)
constexpr int MRGA_OFF = 16384;             // [16][544]  bf16 (aliases dead kvA)
constexpr int H1A_OFF  = 49152;             // [16][192]  bf16 (aliases dead kvA)
constexpr int VOUT_OFF = 174080;            // [160][352] bf16 = 112640
constexpr int QL_OFF   = 286720;            // [8][344]   f32  = 11008
constexpr int SC_OFF   = 297728;            // [8][2][20] f32  = 1280
constexpr int AT_OFF   = 299008;            // [8][2][20] f32  = 1280
constexpr int INV_OFF  = 300288;            // [8] int
constexpr int SMEM_BYTES = 300352;          // <= 320KB/WGP

} // namespace

// A fragment (16x32 bf16, row-major source with element stride `stride`).
// ISA layout: lane r=lane&15, hi=lane>>4; elems 0..7 = K[k0+8*hi..], 8..15 = K[k0+16+8*hi..]
__device__ __forceinline__ v16bf ldA(const bf16* base, int stride, int k0, int lane) {
  const int r = lane & 15, hi = lane >> 4;
  const bf16* p = base + r * stride + k0 + 8 * hi;
  v8bf lo = *(const v8bf*)p;
  v8bf hs = *(const v8bf*)(p + 16);
  return __builtin_shufflevector(lo, hs, 0,1,2,3,4,5,6,7,8,9,10,11,12,13,14,15);
}

// B fragment (32x16 bf16). Weights stored as W[n_out][k] row-major (padded):
// lane n=lane&15 reads elems 0..7 = K[k0+8*hi..], 8..15 = K[k0+16+8*hi..] of row n0+n.
__device__ __forceinline__ v16bf ldB(const bf16* base, int stride, int k0, int lane) {
  const int n = lane & 15, hi = lane >> 4;
  const bf16* p = base + n * stride + k0 + 8 * hi;
  v8bf lo = *(const v8bf*)p;
  v8bf hs = *(const v8bf*)(p + 16);
  return __builtin_shufflevector(lo, hs, 0,1,2,3,4,5,6,7,8,9,10,11,12,13,14,15);
}

__device__ __forceinline__ v8f wmma_bf(v16bf a, v16bf b, v8f c) {
  return __builtin_amdgcn_wmma_f32_16x16x32_bf16(false, a, false, b, (short)0, c,
                                                 false, false);
}

__device__ __forceinline__ v4bf cvt4(float4 v) {
  v4bf o;
  o[0] = (bf16)v.x; o[1] = (bf16)v.y; o[2] = (bf16)v.z; o[3] = (bf16)v.w;
  return o;
}

// ---------------- kernel 1: pad/convert weights to bf16 ----------------
__global__ __launch_bounds__(256) void k_pad(const float* __restrict__ src,
                                             bf16* __restrict__ dst,
                                             int R, int C, int Cp, int total) {
  int i = blockIdx.x * 256 + threadIdx.x;
  if (i >= total) return;
  int r = i / Cp, c = i % Cp;
  float v = (r < R && c < C) ? src[r * C + c] : 0.0f;
  dst[i] = (bf16)v;
}

// ---------------- kernel 2: q = concat(node,time) @ Wq.T + bq ----------------
__global__ __launch_bounds__(256) void k_qproj(const float* __restrict__ node,
                                               const float* __restrict__ timef,
                                               const bf16* __restrict__ WQp,
                                               const float* __restrict__ bq,
                                               float* __restrict__ qout) {
  __shared__ bf16 Asm[64 * EP_];
  const int b0 = blockIdx.x * 64;
  const int t = threadIdx.x;
  // vectorized staging: all segment boundaries (172, 344) are multiples of 4
  for (int u = t; u < 64 * (EP_ / 4); u += 256) {
    int r = u / (EP_ / 4), c = (u % (EP_ / 4)) * 4;
    int gb = b0 + r;
    float4 v = make_float4(0.f, 0.f, 0.f, 0.f);
    if (gb < B_) {
      if (c < F_)      v = *(const float4*)(node + gb * F_ + c);
      else if (c < E_) v = *(const float4*)(timef + gb * F_ + (c - F_));
    }
    *(v4bf*)(Asm + r * EP_ + c) = cvt4(v);
  }
  __syncthreads();
  const int wid = t >> 5, lane = t & 31;
  const int ln = lane & 15, hi = lane >> 4;
  // 4-way M blocking + software-pipelined (double-buffered) B fragments
  for (int nt = wid; nt < 22; nt += 8) {
    int cg = nt * 16 + ln;
    float bias = (cg < E_) ? bq[cg] : 0.0f;
    v8f c[4];
#pragma unroll
    for (int m = 0; m < 4; ++m)
#pragma unroll
      for (int i = 0; i < 8; ++i) c[m][i] = bias;
    const bf16* Bb = WQp + nt * 16 * EP_;
    v16bf bcur = ldB(Bb, EP_, 0, lane);
#pragma unroll
    for (int kc = 0; kc < 11; ++kc) {
      v16bf bnxt;
      if (kc < 10) bnxt = ldB(Bb, EP_, (kc + 1) * 32, lane);
#pragma unroll
      for (int m = 0; m < 4; ++m)
        c[m] = wmma_bf(ldA(Asm + m * 16 * EP_, EP_, kc * 32, lane), bcur, c[m]);
      bcur = bnxt;
    }
    if (cg < E_) {
#pragma unroll
      for (int m = 0; m < 4; ++m)
#pragma unroll
        for (int rr = 0; rr < 8; ++rr) {
          int gb = b0 + m * 16 + 8 * hi + rr;
          if (gb < B_) qout[gb * E_ + cg] = c[m][rr];
        }
    }
  }
}

// ---------------- kernel 3: fused KV-proj + attention + Wo + merge ----------------
__global__ __launch_bounds__(256) void k_attn(
    const float* __restrict__ node, const float* __restrict__ ngh,
    const float* __restrict__ ngt, const float* __restrict__ edge,
    const unsigned char* __restrict__ mask,
    const bf16* __restrict__ WKp, const bf16* __restrict__ WVp,
    const bf16* __restrict__ WOp, const bf16* __restrict__ W1p,
    const bf16* __restrict__ W2p,
    const float* __restrict__ bv, const float* __restrict__ bo,
    const float* __restrict__ b1v, const float* __restrict__ b2v,
    const float* __restrict__ qws, float* __restrict__ out) {
  extern __shared__ char smem[];
  bf16*  kvA  = (bf16*)(smem + KVA_OFF);    // [160][544]
  bf16*  vO   = (bf16*)(smem + VOUT_OFF);   // [160][352]
  float* qL   = (float*)(smem + QL_OFF);    // [8][344]
  float* scL  = (float*)(smem + SC_OFF);    // [8][2][20]
  float* atL  = (float*)(smem + AT_OFF);    // [8][2][20]
  int*   invL = (int*)(smem + INV_OFF);     // [8]
  bf16*  ctxA = (bf16*)(smem + CTXA_OFF);   // [16][352]  (after phase1)
  bf16*  mrgA = (bf16*)(smem + MRGA_OFF);   // [16][544]
  bf16*  h1A  = (bf16*)(smem + H1A_OFF);    // [16][192]

  const int t = threadIdx.x;
  const int bb0 = blockIdx.x * 8;           // 8 batch elements per WG, 12500 WGs
  const int wid = t >> 5, lane = t & 31;
  const int ln = lane & 15, hi = lane >> 4;

  // ---- phase 0: stage kv = concat(ngh, edge, time) as bf16 (float4 path);
  //      stage q; clear scores
  constexpr int KD4 = KDP_ / 4;             // 136
  for (int u = t; u < 160 * KD4; u += 256) {
    int r = u / KD4, c = (u % KD4) * 4;
    int rowg = bb0 * N_ + r;                // global neighbor row
    float4 v = make_float4(0.f, 0.f, 0.f, 0.f);
    if (c < F_)          v = *(const float4*)(ngh + rowg * F_ + c);
    else if (c < 2 * F_) v = *(const float4*)(edge + rowg * F_ + (c - F_));
    else if (c < KD_)    v = *(const float4*)(ngt + rowg * F_ + (c - 2 * F_));
    *(v4bf*)(kvA + r * KDP_ + c) = cvt4(v);
  }
  for (int u = t; u < 8 * (E_ / 4); u += 256) {
    int bl = u / (E_ / 4), c = (u % (E_ / 4)) * 4;
    *(float4*)(qL + bl * E_ + c) = *(const float4*)(qws + (bb0 + bl) * E_ + c);
  }
  for (int idx = t; idx < 8 * 2 * N_; idx += 256) scL[idx] = 0.0f;
  __syncthreads();

  // ---- phase 1: K and V projections with 5-way M register blocking.
  // 88 work units: {V,K} x {mt-group 0/1} x 22 nt = exactly 11 per wave.
  // One (double-buffered) B fragment per k-chunk feeds 5 independent
  // accumulators: 5x less L2 traffic, no C RAW chain, load-ahead-by-one.
  // V tiles -> bf16 LDS. K tiles -> score accumulation (bk dropped: softmax invariant).
  for (int u = wid; u < 88; u += 8) {
    const int matsel = u / 44;              // 0 = V, 1 = K
    const int rest = u % 44;
    const int mt0 = (rest / 22) * 5;        // 0 or 5
    const int nt = rest % 22;
    const int cg = nt * 16 + ln;
    const bool isv = (matsel == 0);
    const bf16* Wp = isv ? WVp : WKp;
    float bias = (isv && cg < E_) ? bv[cg] : 0.0f;
    v8f c[5];
#pragma unroll
    for (int m = 0; m < 5; ++m)
#pragma unroll
      for (int i = 0; i < 8; ++i) c[m][i] = bias;
    const bf16* Bb = Wp + nt * 16 * KDP_;
    v16bf bcur = ldB(Bb, KDP_, 0, lane);
#pragma unroll
    for (int kc = 0; kc < 17; ++kc) {
      v16bf bnxt;
      if (kc < 16) bnxt = ldB(Bb, KDP_, (kc + 1) * 32, lane);
#pragma unroll
      for (int m = 0; m < 5; ++m)
        c[m] = wmma_bf(ldA(kvA + (mt0 + m) * 16 * KDP_, KDP_, kc * 32, lane),
                       bcur, c[m]);
      bcur = bnxt;
    }
    if (isv) {
#pragma unroll
      for (int m = 0; m < 5; ++m)
#pragma unroll
        for (int rr = 0; rr < 8; ++rr) {
          int rg = (mt0 + m) * 16 + 8 * hi + rr;
          vO[rg * EP_ + cg] = (bf16)c[m][rr];
        }
    } else if (cg < E_) {
      const int hh = (cg >= F_) ? 1 : 0;
#pragma unroll
      for (int m = 0; m < 5; ++m)
#pragma unroll
        for (int rr = 0; rr < 8; ++rr) {
          int rg = (mt0 + m) * 16 + 8 * hi + rr;
          int bl = rg / N_, nn = rg % N_;
          atomicAdd(&scL[bl * 40 + hh * N_ + nn], c[m][rr] * qL[bl * E_ + cg]);
        }
    }
  }
  __syncthreads();

  // ---- phase 2: masked softmax per (b,h); handle all-masked rows
  if (t < 16) {
    int bl = t >> 1, hh = t & 1;
    const unsigned char* mrow = mask + (size_t)(bb0 + bl) * N_;
    bool allm = true;
    for (int n = 0; n < N_; ++n) allm = allm && (mrow[n] != 0);
    if (hh == 0) invL[bl] = allm ? 1 : 0;
    float mx = -1e30f;
    for (int n = 0; n < N_; ++n) {
      bool msk = (mrow[n] != 0) && !(allm && n == 0);
      if (!msk) mx = fmaxf(mx, scL[bl * 40 + hh * N_ + n] * SCALE_);
    }
    float sum = 0.0f;
    for (int n = 0; n < N_; ++n) {
      bool msk = (mrow[n] != 0) && !(allm && n == 0);
      float ev = msk ? 0.0f : __expf(scL[bl * 40 + hh * N_ + n] * SCALE_ - mx);
      atL[bl * 40 + hh * N_ + n] = ev;
      sum += ev;
    }
    float inv = 1.0f / sum;
    for (int n = 0; n < N_; ++n) atL[bl * 40 + hh * N_ + n] *= inv;
  }
  __syncthreads();

  // ---- phase 3: attn_w output, ctx (VALU), and prefills for merge tiles
  for (int idx = t; idx < 8 * N_; idx += 256) {
    int bl = idx / N_, n = idx % N_;
    float aw = invL[bl] ? 0.0f : 0.5f * (atL[bl * 40 + n] + atL[bl * 40 + N_ + n]);
    out[(size_t)B_ * F_ + (size_t)(bb0 + bl) * N_ + n] = aw;
  }
  for (int idx = t; idx < 16 * EP_; idx += 256) {   // ctxA: full 16x352, zero padding
    int r = idx / EP_, e = idx % EP_;
    float acc = 0.0f;
    if (r < 8 && e < E_) {
      int hh = (e >= F_) ? 1 : 0;
      const float* at = &atL[r * 40 + hh * N_];
      const bf16* vrow = vO + r * N_ * EP_ + e;
      for (int n = 0; n < N_; ++n) acc += at[n] * (float)vrow[n * EP_];
    }
    ctxA[idx] = (bf16)acc;
  }
  for (int idx = t; idx < 16 * KDP_; idx += 256) {  // mergeA: src_node cols + zero pad
    int r = idx / KDP_, c = idx % KDP_;
    if (r < 8 && c < E_) continue;                  // written by phase 4 WMMA
    float v = 0.0f;
    if (r < 8 && c >= E_ && c < KD_) v = node[(bb0 + r) * F_ + (c - E_)];
    mrgA[idx] = (bf16)v;
  }
  for (int idx = t; idx < 16 * H2P_; idx += 256) {  // h1A: zero padding regions
    int r = idx / H2P_, c = idx % H2P_;
    if (r >= 8 || c >= HP_) h1A[idx] = (bf16)0.0f;
  }
  __syncthreads();

  // ---- phase 4: attn_out = ctx @ Wo.T + bo  (M=8 padded to 16), into mergeA
  for (int nt = wid; nt < 22; nt += 8) {
    int cg = nt * 16 + ln;
    float bias = (cg < E_) ? bo[cg] : 0.0f;
    v8f c;
#pragma unroll
    for (int i = 0; i < 8; ++i) c[i] = bias;
    const bf16* Bb = WOp + nt * 16 * EP_;
    v16bf bcur = ldB(Bb, EP_, 0, lane);
#pragma unroll
    for (int kc = 0; kc < 11; ++kc) {
      v16bf bnxt;
      if (kc < 10) bnxt = ldB(Bb, EP_, (kc + 1) * 32, lane);
      c = wmma_bf(ldA(ctxA, EP_, kc * 32, lane), bcur, c);
      bcur = bnxt;
    }
    if (hi == 0 && cg < E_) {
#pragma unroll
      for (int rr = 0; rr < 8; ++rr) {
        float v = invL[rr] ? 0.0f : c[rr];
        mrgA[rr * KDP_ + cg] = (bf16)v;
      }
    }
  }
  __syncthreads();

  // ---- phase 5: h1 = relu(mergeA @ W1.T + b1)
  for (int nt = wid; nt < 11; nt += 8) {
    int cg = nt * 16 + ln;
    float bias = (cg < HID_) ? b1v[cg] : 0.0f;
    v8f c;
#pragma unroll
    for (int i = 0; i < 8; ++i) c[i] = bias;
    const bf16* Bb = W1p + nt * 16 * KDP_;
    v16bf bcur = ldB(Bb, KDP_, 0, lane);
#pragma unroll
    for (int kc = 0; kc < 17; ++kc) {
      v16bf bnxt;
      if (kc < 16) bnxt = ldB(Bb, KDP_, (kc + 1) * 32, lane);
      c = wmma_bf(ldA(mrgA, KDP_, kc * 32, lane), bcur, c);
      bcur = bnxt;
    }
    if (hi == 0) {
#pragma unroll
      for (int rr = 0; rr < 8; ++rr)
        h1A[rr * H2P_ + cg] = (bf16)fmaxf(c[rr], 0.0f);
    }
  }
  __syncthreads();

  // ---- phase 6: out = h1 @ W2.T + b2 -> global
  for (int nt = wid; nt < 11; nt += 8) {
    int cg = nt * 16 + ln;
    float bias = (cg < F_) ? b2v[cg] : 0.0f;
    v8f c;
#pragma unroll
    for (int i = 0; i < 8; ++i) c[i] = bias;
    const bf16* Bb = W2p + nt * 16 * H2P_;
    v16bf bcur = ldB(Bb, H2P_, 0, lane);
#pragma unroll
    for (int kc = 0; kc < 6; ++kc) {
      v16bf bnxt;
      if (kc < 5) bnxt = ldB(Bb, H2P_, (kc + 1) * 32, lane);
      c = wmma_bf(ldA(h1A, H2P_, kc * 32, lane), bcur, c);
      bcur = bnxt;
    }
    if (hi == 0 && cg < F_) {
#pragma unroll
      for (int rr = 0; rr < 8; ++rr)
        out[(size_t)(bb0 + rr) * F_ + cg] = c[rr];
    }
  }
}

extern "C" void kernel_launch(void* const* d_in, const int* in_sizes, int n_in,
                              void* d_out, int out_size, void* d_ws, size_t ws_size,
                              hipStream_t stream) {
  (void)in_sizes; (void)n_in; (void)out_size; (void)ws_size;
  const float* node  = (const float*)d_in[0];
  const float* timef = (const float*)d_in[1];
  const float* ngh   = (const float*)d_in[2];
  const float* ngt   = (const float*)d_in[3];
  const float* edge  = (const float*)d_in[4];
  const unsigned char* mask = (const unsigned char*)d_in[5];
  const float* Wq = (const float*)d_in[6];
  const float* Wk = (const float*)d_in[7];
  const float* Wv = (const float*)d_in[8];
  const float* bq = (const float*)d_in[9];
  // bk (d_in[10]) intentionally unused: uniform key bias is softmax-invariant
  const float* bv = (const float*)d_in[11];
  const float* Wo = (const float*)d_in[12];
  const float* bo = (const float*)d_in[13];
  const float* W1 = (const float*)d_in[14];
  const float* b1 = (const float*)d_in[15];
  const float* W2 = (const float*)d_in[16];
  const float* b2 = (const float*)d_in[17];
  float* out = (float*)d_out;

  char* ws = (char*)d_ws;
  bf16* WQp = (bf16*)(ws + WQ_OFF);
  bf16* WKp = (bf16*)(ws + WK_OFF);
  bf16* WVp = (bf16*)(ws + WV_OFF);
  bf16* WOp = (bf16*)(ws + WO_OFF);
  bf16* W1p = (bf16*)(ws + W1_OFF);
  bf16* W2p = (bf16*)(ws + W2_OFF);
  float* qws = (float*)(ws + QW_OFF);

  auto g = [](int total) { return dim3((unsigned)((total + 255) / 256)); };
  k_pad<<<g(EP_ * EP_),  256, 0, stream>>>(Wq, WQp, E_,  E_,  EP_,  EP_ * EP_);
  k_pad<<<g(EP_ * KDP_), 256, 0, stream>>>(Wk, WKp, E_,  KD_, KDP_, EP_ * KDP_);
  k_pad<<<g(EP_ * KDP_), 256, 0, stream>>>(Wv, WVp, E_,  KD_, KDP_, EP_ * KDP_);
  k_pad<<<g(EP_ * EP_),  256, 0, stream>>>(Wo, WOp, E_,  E_,  EP_,  EP_ * EP_);
  k_pad<<<g(HP_ * KDP_), 256, 0, stream>>>(W1, W1p, HID_, KD_, KDP_, HP_ * KDP_);
  k_pad<<<g(HP_ * H2P_), 256, 0, stream>>>(W2, W2p, HID_, HID_, H2P_, HP_ * H2P_);

  k_qproj<<<dim3((B_ + 63) / 64), 256, 0, stream>>>(node, timef, WQp, bq, qws);

  (void)hipFuncSetAttribute((const void*)k_attn,
                            hipFuncAttributeMaxDynamicSharedMemorySize, SMEM_BYTES);
  k_attn<<<dim3(B_ / 8), 256, SMEM_BYTES, stream>>>(
      node, ngh, ngt, edge, mask, WKp, WVp, WOp, W1p, W2p,
      bv, bo, b1, b2, qws, out);
}
